// AudioModelX2_88424786690500
// MI455X (gfx1250) — matrix operand order repaired and checked by reading, not compiled, not run on hardware
//
#include <hip/hip_runtime.h>
#include <math.h>

// ---------------------------------------------------------------------------
// CDNA5 (gfx1250) xLSTM audio model. bf16 WMMA (f32 accum) for all deep
// contractions; TDM + async global->LDS staging in the dense GEMM;
// VALU kernels for skinny/elementwise ops.
// Workspace requirement: ~537 MB of float scratch in d_ws.
// ---------------------------------------------------------------------------

typedef __bf16 bf16;
typedef __attribute__((ext_vector_type(16))) __bf16 bf16x16;
typedef __attribute__((ext_vector_type(8)))  float  floatx8;
typedef __attribute__((ext_vector_type(4)))  unsigned u32x4;
typedef __attribute__((ext_vector_type(8)))  unsigned u32x8;

#define WMMA_BF16(a, b, c) \
  __builtin_amdgcn_wmma_f32_16x16x32_bf16(false, (a), false, (b), (short)0, (c), false, false)

#define NBATCH 8
#define SEQ    1024
#define DMODEL 1024
#define BSROWS (NBATCH * SEQ)   // 8192
#define INNER  2048
#define NHEAD  4
#define DH_M   512
#define DH_S   256
#define FFD    1344

// ============================ generic WMMA GEMM ============================
// C[m,n] = sum_k A[m,k] * B[n,k]  (+bias[n]) (+res[m,n])
// 64x64 output tile per block, K step 32; 128 threads = 4 waves; wave w owns
// rows [16w,16w+16) x all 64 cols. Tile staging uses the CDNA5 async engine:
//   A tile: one TENSOR_LOAD_TO_LDS 2D descriptor (wave 0), TENSORcnt
//   B tile: per-lane GLOBAL_LOAD_ASYNC_TO_LDS_B128, ASYNCcnt
// LDS holds fp32; fp32->bf16 conversion happens at fragment build (cvt_pk
// co-executes with WMMA). Dims must be multiples of 64 (M,N) / 32 (K).
__global__ __launch_bounds__(128)
void gemm_bf16w(const float* __restrict__ Abase, const float* __restrict__ Bbase,
                const float* __restrict__ bias,  const float* __restrict__ resBase,
                float* __restrict__ Cbase,
                int M, int N, int K, int lda, int ldb, int ldc,
                long aB, long bB, long rB, long cB)
{
  const float* A = Abase + (long)blockIdx.z * aB;
  const float* Bw = Bbase + (long)blockIdx.z * bB;
  const float* R = resBase ? resBase + (long)blockIdx.z * rB : nullptr;
  float* C = Cbase + (long)blockIdx.z * cB;

  __shared__ float Asf[64][32];   // 8 KB
  __shared__ float Bsf[64][32];   // 8 KB

  const int tid = threadIdx.x;
  const int wave = tid >> 5, lane = tid & 31;
  const int ml = lane & 15, hs = lane >> 4;
  const int m0 = blockIdx.x * 64, n0 = blockIdx.y * 64;

  // Generic LDS pointers carry the LDS byte offset in their low 32 bits
  // (flat aperture rule).
  const unsigned ldsA = (unsigned)(unsigned long long)(&Asf[0][0]);
  const unsigned ldsB = (unsigned)(unsigned long long)(&Bsf[0][0]);

  floatx8 acc[4] = {};
  for (int k0 = 0; k0 < K; k0 += 32) {
    __syncthreads();

    // ---- A tile: Tensor Data Mover, 64 rows x 32 fp32, row stride lda ----
    if (wave == 0) {
      unsigned long long ga = (unsigned long long)(A + (size_t)m0 * lda + k0);
      u32x4 g0; u32x8 g1;
      g0[0] = 1u;                                  // count=1, user descriptor
      g0[1] = ldsA;                                // lds_addr (bytes)
      g0[2] = (unsigned)ga;                        // global_addr[31:0]
      g0[3] = (unsigned)(ga >> 32) | (2u << 30);   // global_addr[56:32] | type=2
      g1[0] = 0x20000u;                            // data_size = 4 bytes
      g1[1] = 32u << 16;                           // tensor_dim0 = 32
      g1[2] = 64u << 16;                           // tensor_dim1 = 64
      g1[3] = 32u << 16;                           // tile_dim0 = 32
      g1[4] = 64u;                                 // tile_dim1 = 64
      g1[5] = (unsigned)lda;                       // tensor_dim0_stride (elems)
      g1[6] = 0u; g1[7] = 0u;
      asm volatile("tensor_load_to_lds %0, %1" :: "s"(g0), "s"(g1) : "memory");
    }

    // ---- B tile: per-lane async global->LDS (4 x b128 per thread) ----
    for (int idx = tid; idx < 64 * 8; idx += 128) {
      int r = idx >> 3, c4 = (idx & 7) << 2;
      const float* src = Bw + (size_t)(n0 + r) * ldb + k0 + c4;
      unsigned dst = ldsB + (unsigned)((r * 32 + c4) << 2);
      asm volatile("global_load_async_to_lds_b128 %0, %1, off"
                   :: "v"(dst), "v"(src) : "memory");
    }

    asm volatile("s_wait_asynccnt 0" ::: "memory");
    if (wave == 0) __builtin_amdgcn_s_wait_tensorcnt(0);
    __syncthreads();

    bf16x16 af;
    #pragma unroll
    for (int e = 0; e < 8; ++e) af[e] = (bf16)Asf[wave * 16 + ml][16 * hs + e];
    #pragma unroll
    for (int e = 0; e < 8; ++e) af[8 + e] = (bf16)Asf[wave * 16 + ml][16 * hs + 8 + e];
    #pragma unroll
    for (int j = 0; j < 4; ++j) {
      bf16x16 bfm;
      #pragma unroll
      for (int e = 0; e < 16; ++e) bfm[e] = (bf16)Bsf[j * 16 + ml][16 * hs + e];
      acc[j] = WMMA_BF16(af, bfm, acc[j]);
    }
  }
  #pragma unroll
  for (int j = 0; j < 4; ++j) {
    #pragma unroll
    for (int r = 0; r < 8; ++r) {
      int row = m0 + wave * 16 + r + 8 * hs;
      int col = n0 + j * 16 + ml;
      float v = acc[j][r];
      if (bias) v += bias[col];
      if (R) v += R[(size_t)row * ldc + col];
      C[(size_t)row * ldc + col] = v;
    }
  }
}

// =========================== LN / group-norm ===============================
__global__ __launch_bounds__(256)
void rownorm(const float* __restrict__ x, const float* __restrict__ w,
             float* __restrict__ y, int Dc, int wMod, float eps)
{
  long row = blockIdx.x;
  int tid = threadIdx.x;
  __shared__ float r1[256], r2[256];
  const float* xr = x + row * Dc;
  float s1 = 0.f, s2 = 0.f;
  for (int c = tid; c < Dc; c += 256) { float v = xr[c]; s1 += v; s2 += v * v; }
  r1[tid] = s1; r2[tid] = s2; __syncthreads();
  for (int off = 128; off > 0; off >>= 1) {
    if (tid < off) { r1[tid] += r1[tid + off]; r2[tid] += r2[tid + off]; }
    __syncthreads();
  }
  float mu = r1[0] / Dc, var = r2[0] / Dc - mu * mu;
  float ri = rsqrtf(var + eps);
  const float* wr = w + (long)(row % wMod) * Dc;
  float* yr = y + row * Dc;
  for (int c = tid; c < Dc; c += 256) yr[c] = (xr[c] - mu) * ri * wr[c];
}

// ======================= causal depthwise conv + SiLU ======================
__global__ void conv_silu(const float* __restrict__ x, const float* __restrict__ w,
                          const float* __restrict__ bia, float* __restrict__ y,
                          int C, int ldx, int ldy)
{
  long i = blockIdx.x * (long)blockDim.x + threadIdx.x;
  if (i >= (long)BSROWS * C) return;
  int c = (int)(i % C); long bs = i / C; int s = (int)(bs % SEQ);
  float a = bia[c];
  #pragma unroll
  for (int k = 0; k < 4; ++k) {
    int sp = s - 3 + k;
    if (sp >= 0) a += x[(bs - s + sp) * (long)ldx + c] * w[c * 4 + k];
  }
  y[bs * (long)ldy + c] = a / (1.f + __expf(-a));
}

// ===================== 4x4 block-diagonal projections ======================
__global__ void headwise4(const float* __restrict__ x, const float* __restrict__ w,
                          float* __restrict__ y, int ldx, int ldy, int colOff)
{
  long i = blockIdx.x * (long)blockDim.x + threadIdx.x;
  if (i >= (long)BSROWS * INNER) return;
  int c = (int)(i & (INNER - 1)); long bs = i >> 11;
  int n = c >> 2, o = c & 3;
  const float* xr = x + bs * (long)ldx + (n << 2);
  const float* wr = w + n * 16 + o * 4;
  float a = xr[0] * wr[0] + xr[1] * wr[1] + xr[2] * wr[2] + xr[3] * wr[3];
  y[bs * (long)ldy + colOff + c] = a;
}

// ====================== ig / fg gate dot products ==========================
__global__ __launch_bounds__(256)
void igfg_kernel(const float* __restrict__ qkv,
                 const float* __restrict__ igw, const float* __restrict__ igb,
                 const float* __restrict__ fgw, const float* __restrict__ fgb,
                 float* __restrict__ ig, float* __restrict__ fg)
{
  long bs = blockIdx.x;
  int wave = threadIdx.x >> 5, lane = threadIdx.x & 31;
  int n = wave & 3; bool isF = wave >= 4;
  const float* wrow = (isF ? fgw : igw) + n * (3 * INNER);
  const float* xr = qkv + bs * (long)(3 * INNER);
  float s = 0.f;
  for (int k = lane; k < 3 * INNER; k += 32) s += xr[k] * wrow[k];
  #pragma unroll
  for (int off = 16; off > 0; off >>= 1) s += __shfl_xor(s, off, 32);
  if (lane == 0) {
    long b = bs / SEQ, si = bs % SEQ;
    float v = s + (isF ? fgb[n] : igb[n]);
    (isF ? fg : ig)[(b * NHEAD + n) * SEQ + si] = v;
  }
}

// ================= cumsum of log_sigmoid(fg) over sequence =================
__global__ __launch_bounds__(1024)
void lsig_scan(const float* __restrict__ fg, float* __restrict__ lfc)
{
  __shared__ float buf[2][1024];
  int row = blockIdx.x, t = threadIdx.x;
  float x = fg[(long)row * SEQ + t];
  float ls = fminf(x, 0.f) - log1pf(__expf(-fabsf(x)));
  buf[0][t] = ls; __syncthreads();
  int src = 0;
  for (int off = 1; off < 1024; off <<= 1) {
    float v = buf[src][t] + ((t >= off) ? buf[src][t - off] : 0.f);
    buf[src ^ 1][t] = v; __syncthreads(); src ^= 1;
  }
  lfc[(long)row * SEQ + t] = buf[src][t];
}

// ================= fused streaming stabilized mLSTM attention ==============
// block = (q-tile of 16 rows, head n, batch b); 128 threads = 4 waves, wave w
// owns DH columns [128w,128w+128). Streams 32-wide t-tiles with running
// row-max/row-sum rescaling; WMMA for QK^T and PV.
__global__ __launch_bounds__(128)
void mlstm_attn(const float* __restrict__ qkv, const float* __restrict__ ig,
                const float* __restrict__ lfc, float* __restrict__ hout)
{
  const int b = blockIdx.z, n = blockIdx.y, q0 = blockIdx.x * 16;
  const int tid = threadIdx.x, wave = tid >> 5, lane = tid & 31;
  const int ml = lane & 15, hs = lane >> 4;

  __shared__ bf16 qs[16][DH_M];      // 16 KB
  __shared__ bf16 ks[32][128];       //  8 KB
  __shared__ bf16 vs[32][DH_M];      // 32 KB
  __shared__ bf16 cb[4][16][32];     //  4 KB (per-wave P tiles)
  __shared__ float gt_lfc[32], gt_ig[32];

  // stage q tile (head n lives at qkv col n*512 within the q block)
  for (int idx = tid; idx < 16 * (DH_M / 4); idx += 128) {
    int r = idx / (DH_M / 4), c4 = (idx % (DH_M / 4)) * 4;
    float4 v = *(const float4*)(qkv + ((long)(b * SEQ + q0 + r)) * (3 * INNER) + n * DH_M + c4);
    qs[r][c4 + 0] = (bf16)v.x; qs[r][c4 + 1] = (bf16)v.y;
    qs[r][c4 + 2] = (bf16)v.z; qs[r][c4 + 3] = (bf16)v.w;
  }

  const long gbase = ((long)b * NHEAD + n) * SEQ;
  float lfcr[8], mrun[8], srun[8];
  floatx8 acc[8] = {};
  #pragma unroll
  for (int r = 0; r < 8; ++r) {
    lfcr[r] = lfc[gbase + q0 + r + 8 * hs];
    mrun[r] = -3.0e38f; srun[r] = 0.f;
  }

  const float scale = 0.044194173824159216f;  // 512^-0.5

  for (int t0 = 0; t0 < q0 + 16; t0 += 32) {
    __syncthreads();
    // stage v tile (32 x 512)
    for (int idx = tid; idx < 32 * (DH_M / 4); idx += 128) {
      int r = idx / (DH_M / 4), c4 = (idx % (DH_M / 4)) * 4;
      float4 v = *(const float4*)(qkv + ((long)(b * SEQ + t0 + r)) * (3 * INNER)
                                  + 2 * INNER + n * DH_M + c4);
      vs[r][c4 + 0] = (bf16)v.x; vs[r][c4 + 1] = (bf16)v.y;
      vs[r][c4 + 2] = (bf16)v.z; vs[r][c4 + 3] = (bf16)v.w;
    }
    if (tid < 32) { gt_lfc[tid] = lfc[gbase + t0 + tid]; gt_ig[tid] = ig[gbase + t0 + tid]; }

    // ---- scores: C(16x32) = q @ k^T over DH=512, chunked by 128 ----
    floatx8 sc0 = {}, sc1 = {};
    for (int kc = 0; kc < 4; ++kc) {
      __syncthreads();
      for (int idx = tid; idx < 32 * 32; idx += 128) {     // k chunk 32x128
        int r = idx >> 5, c4 = (idx & 31) * 4;
        float4 v = *(const float4*)(qkv + ((long)(b * SEQ + t0 + r)) * (3 * INNER)
                                    + INNER + n * DH_M + kc * 128 + c4);
        ks[r][c4 + 0] = (bf16)v.x; ks[r][c4 + 1] = (bf16)v.y;
        ks[r][c4 + 2] = (bf16)v.z; ks[r][c4 + 3] = (bf16)v.w;
      }
      __syncthreads();
      #pragma unroll
      for (int kk = 0; kk < 4; ++kk) {
        const int kb = kc * 128 + kk * 32;
        bf16x16 af;
        #pragma unroll
        for (int e = 0; e < 8; ++e) af[e] = qs[ml][kb + 16 * hs + e];
        #pragma unroll
        for (int e = 0; e < 8; ++e) af[8 + e] = qs[ml][kb + 16 * hs + 8 + e];
        bf16x16 b0, b1;
        #pragma unroll
        for (int e = 0; e < 16; ++e) b0[e] = ks[ml][kk * 32 + 16 * hs + e];
        #pragma unroll
        for (int e = 0; e < 16; ++e) b1[e] = ks[16 + ml][kk * 32 + 16 * hs + e];
        sc0 = WMMA_BF16(af, b0, sc0);
        sc1 = WMMA_BF16(af, b1, sc1);
      }
    }

    // ---- log-D weights, running max, rescale, exp ----
    float rsf[8];
    #pragma unroll
    for (int r = 0; r < 8; ++r) {
      int srow = q0 + r + 8 * hs;
      int t0c = t0 + ml, t1c = t0 + 16 + ml;
      float w0 = (t0c <= srow) ? (lfcr[r] - gt_lfc[ml] + gt_ig[ml]) : -INFINITY;
      float w1 = (t1c <= srow) ? (lfcr[r] - gt_lfc[16 + ml] + gt_ig[16 + ml]) : -INFINITY;
      float wmax = fmaxf(w0, w1);
      #pragma unroll
      for (int off = 1; off < 16; off <<= 1) wmax = fmaxf(wmax, __shfl_xor(wmax, off, 32));
      float mn = fmaxf(mrun[r], wmax);
      rsf[r] = __expf(mrun[r] - mn);
      float e0 = (w0 == -INFINITY) ? 0.f : __expf(w0 - mn);
      float e1 = (w1 == -INFINITY) ? 0.f : __expf(w1 - mn);
      float p0 = sc0[r] * scale * e0;
      float p1 = sc1[r] * scale * e1;
      float rsum = p0 + p1;
      #pragma unroll
      for (int off = 1; off < 16; off <<= 1) rsum += __shfl_xor(rsum, off, 32);
      srun[r] = srun[r] * rsf[r] + rsum;
      mrun[r] = mn;
      cb[wave][r + 8 * hs][ml]      = (bf16)p0;
      cb[wave][r + 8 * hs][16 + ml] = (bf16)p1;
    }
    #pragma unroll
    for (int j = 0; j < 8; ++j)
      #pragma unroll
      for (int r = 0; r < 8; ++r) acc[j][r] *= rsf[r];

    // ---- PV: acc(16x128) += P(16x32) @ v(32x128 for this wave) ----
    bf16x16 a2;
    #pragma unroll
    for (int e = 0; e < 8; ++e) a2[e] = cb[wave][ml][16 * hs + e];
    #pragma unroll
    for (int e = 0; e < 8; ++e) a2[8 + e] = cb[wave][ml][16 * hs + 8 + e];
    #pragma unroll
    for (int j = 0; j < 8; ++j) {
      bf16x16 bv;
      #pragma unroll
      for (int e = 0; e < 16; ++e) bv[e] = vs[16 * hs + e][wave * 128 + j * 16 + ml];
      acc[j] = WMMA_BF16(a2, bv, acc[j]);
    }
  }

  // ---- normalize & store (B,S,NH,DH) ----
  float invn[8];
  #pragma unroll
  for (int r = 0; r < 8; ++r) {
    float nrm = fmaxf(fabsf(srun[r]), __expf(-mrun[r])) + 1e-6f;
    invn[r] = 1.f / nrm;
  }
  #pragma unroll
  for (int j = 0; j < 8; ++j)
    #pragma unroll
    for (int r = 0; r < 8; ++r) {
      int srow = q0 + r + 8 * hs;
      int col = wave * 128 + j * 16 + ml;
      hout[(((long)b * SEQ + srow) * NHEAD + n) * DH_M + col] = acc[j][r] * invn[r];
    }
}

// ===================== mLSTM combine: (h+skip*xc)*silu(z) ==================
__global__ void combine_mlstm(float* __restrict__ h, const float* __restrict__ xc,
                              const float* __restrict__ up, const float* __restrict__ skip)
{
  long i = blockIdx.x * (long)blockDim.x + threadIdx.x;
  if (i >= (long)BSROWS * INNER) return;
  int c = (int)(i & (INNER - 1)); long bs = i >> 11;
  float z = up[bs * (long)(2 * INNER) + INNER + c];
  float sz = z / (1.f + __expf(-z));
  h[i] = (h[i] + skip[c] * xc[i]) * sz;
}

// ===================== sLSTM sequential scan (WMMA recurrence) =============
// One block per head, 4 waves (one per gate). Recurrent h kept in LDS as a
// zero-padded 16x256 bf16 A-matrix; R streamed from L2 each step.
__global__ __launch_bounds__(128)
void slstm_scan(const float* __restrict__ gates,  // (4,B,S,D)
                const float* __restrict__ R,      // (4,NH,256,256) [g][n][i][o]
                const float* __restrict__ bias,   // (4,NH,256)
                float* __restrict__ hout)         // (B,S,NH*256)
{
  const int head = blockIdx.x;
  const int tid = threadIdx.x, wave = tid >> 5, lane = tid & 31;
  const int ml = lane & 15, hs = lane >> 4;
  __shared__ bf16 hbf[16][DH_S];       // 8 KB (rows 8..15 zero pad)
  __shared__ float rec[4][8][DH_S];    // 32 KB

  for (int idx = tid; idx < 16 * DH_S; idx += 128) hbf[idx >> 8][idx & 255] = (bf16)0.f;
  float cst[16], nst[16], mst[16];
  #pragma unroll
  for (int e = 0; e < 16; ++e) { cst[e] = 0.f; nst[e] = 0.f; mst[e] = 0.f; }
  __syncthreads();

  const float* Rg = R + ((long)wave * NHEAD + head) * DH_S * DH_S;
  const long BSD = (long)NBATCH * SEQ * DMODEL;

  for (int s = 0; s < SEQ; ++s) {
    #pragma unroll 1
    for (int nt = 0; nt < 16; ++nt) {
      floatx8 a = {};
      #pragma unroll
      for (int kk = 0; kk < 8; ++kk) {
        bf16x16 af;
        #pragma unroll
        for (int e = 0; e < 8; ++e) af[e] = hbf[ml][kk * 32 + 16 * hs + e];
        #pragma unroll
        for (int e = 0; e < 8; ++e) af[8 + e] = hbf[ml][kk * 32 + 16 * hs + 8 + e];
        bf16x16 bfr;
        #pragma unroll
        for (int e = 0; e < 16; ++e)
          bfr[e] = (bf16)Rg[(kk * 32 + 16 * hs + e) * DH_S + nt * 16 + ml];
        a = WMMA_BF16(af, bfr, a);
      }
      #pragma unroll
      for (int r = 0; r < 8; ++r) {
        int row = r + 8 * hs;
        if (row < 8) rec[wave][row][nt * 16 + ml] = a[r];
      }
    }
    __syncthreads();
    #pragma unroll
    for (int e = 0; e < 16; ++e) {
      int slot = tid * 16 + e;
      int bb = slot >> 8, o = slot & 255;
      long gidx = ((long)bb * SEQ + s) * DMODEL + head * DH_S + o;
      float ir = gates[0 * BSD + gidx] + rec[0][bb][o] + bias[(0 * NHEAD + head) * DH_S + o];
      float fr = gates[1 * BSD + gidx] + rec[1][bb][o] + bias[(1 * NHEAD + head) * DH_S + o];
      float zr = gates[2 * BSD + gidx] + rec[2][bb][o] + bias[(2 * NHEAD + head) * DH_S + o];
      float og = gates[3 * BSD + gidx] + rec[3][bb][o] + bias[(3 * NHEAD + head) * DH_S + o];
      float mn = fmaxf(fr + mst[e], ir);
      float iv = __expf(ir - mn);
      float fv = __expf(fr + mst[e] - mn);
      float cn = fv * cst[e] + iv * tanhf(zr);
      float nn = fv * nst[e] + iv;
      float hv = (1.f / (1.f + __expf(-og))) * cn / nn;
      cst[e] = cn; nst[e] = nn; mst[e] = mn;
      hbf[bb][o] = (bf16)hv;
      hout[((long)bb * SEQ + s) * DMODEL + head * DH_S + o] = hv;
    }
    __syncthreads();
  }
}

// =============================== misc elementwise ==========================
__global__ void add_inplace(float* __restrict__ a, const float* __restrict__ b, long n) {
  long i = blockIdx.x * (long)blockDim.x + threadIdx.x;
  if (i < n) a[i] += b[i];
}

__global__ void gelu_gate(const float* __restrict__ up, float* __restrict__ mid) {
  long i = blockIdx.x * (long)blockDim.x + threadIdx.x;
  if (i >= (long)BSROWS * FFD) return;
  int c = (int)(i % FFD); long bs = i / FFD;
  float g = up[bs * (long)(2 * FFD) + c];
  float u = up[bs * (long)(2 * FFD) + FFD + c];
  float t = 0.7978845608028654f * (g + 0.044715f * g * g * g);
  mid[i] = 0.5f * g * (1.f + tanhf(t)) * u;
}

__global__ void selu_mean(const float* __restrict__ x, float* __restrict__ feat) {
  int i = blockIdx.x * blockDim.x + threadIdx.x;
  if (i >= NBATCH * DMODEL) return;
  int b = i / DMODEL, d = i % DMODEL;
  const float al = 1.6732632423543772f, sc = 1.0507009873554805f;
  float s = 0.f;
  for (int t = 0; t < SEQ; ++t) {
    float v = x[((long)b * SEQ + t) * DMODEL + d];
    s += (v > 0.f) ? sc * v : sc * al * (__expf(v) - 1.f);
  }
  feat[i] = s * (1.f / SEQ);
}

__global__ __launch_bounds__(32)
void heads_kernel(const float* __restrict__ feat,
                  const float* __restrict__ ew, const float* __restrict__ eb,
                  const float* __restrict__ sw, const float* __restrict__ sb,
                  float* __restrict__ out)
{
  int o = blockIdx.x, lane = threadIdx.x;
  const float* wrow; float bias; int b;
  if (o < 56) { b = o / 7; int j = o % 7; wrow = ew + j * DMODEL; bias = eb[j]; }
  else        { int q = o - 56; b = q / 3; int j = q % 3; wrow = sw + j * DMODEL; bias = sb[j]; }
  float s = 0.f;
  for (int k = lane; k < DMODEL; k += 32) s += feat[b * DMODEL + k] * wrow[k];
  #pragma unroll
  for (int off = 16; off > 0; off >>= 1) s += __shfl_xor(s, off, 32);
  if (lane == 0) out[o] = s + bias;
}

// ================================ host side ================================
static const size_t OFF_RES  = 0;
static const size_t OFF_LN   = OFF_RES  + (size_t)BSROWS * DMODEL;
static const size_t OFF_UP   = OFF_LN   + (size_t)BSROWS * DMODEL;
static const size_t OFF_XC   = OFF_UP   + (size_t)BSROWS * 2 * INNER;
static const size_t OFF_QKV  = OFF_XC   + (size_t)BSROWS * INNER;
static const size_t OFF_H    = OFF_QKV  + (size_t)BSROWS * 3 * INNER;
static const size_t OFF_IG   = OFF_H    + (size_t)BSROWS * INNER;
static const size_t OFF_FG   = OFF_IG   + (size_t)NBATCH * NHEAD * SEQ;
static const size_t OFF_LFC  = OFF_FG   + (size_t)NBATCH * NHEAD * SEQ;
static const size_t OFF_FEAT = OFF_LFC  + (size_t)NBATCH * NHEAD * SEQ;

extern "C" void kernel_launch(void* const* d_in, const int* in_sizes, int n_in,
                              void* d_out, int out_size, void* d_ws, size_t ws_size,
                              hipStream_t stream)
{
  (void)in_sizes; (void)n_in; (void)out_size; (void)ws_size;
  auto F = [&](int i) { return (const float*)d_in[i]; };
  const float* X = F(0);
  float* ws   = (float*)d_ws;
  float* RES  = ws + OFF_RES;  float* LN  = ws + OFF_LN;  float* UP  = ws + OFF_UP;
  float* XC   = ws + OFF_XC;   float* QKV = ws + OFF_QKV; float* H   = ws + OFF_H;
  float* IG   = ws + OFF_IG;   float* FG  = ws + OFF_FG;  float* LFC = ws + OFF_LFC;
  float* FEAT = ws + OFF_FEAT;
  float* GATES = QKV;  // reuse (needs 4*BS*D <= BS*6144)

  auto mblock = [&](int p0, const float* resIn, float* resOut) {
    const float *ln = F(p0 + 0), *Wup = F(p0 + 1), *cw = F(p0 + 2), *cbp = F(p0 + 3),
                *qw = F(p0 + 4), *kw = F(p0 + 5), *vw = F(p0 + 6),
                *igw = F(p0 + 7), *igb = F(p0 + 8), *fgw = F(p0 + 9), *fgb = F(p0 + 10),
                *nw = F(p0 + 11), *sk = F(p0 + 12), *Wd = F(p0 + 13);
    rownorm<<<BSROWS, 256, 0, stream>>>(resIn, ln, LN, DMODEL, 1, 1e-5f);
    gemm_bf16w<<<dim3(BSROWS / 64, (2 * INNER) / 64, 1), 128, 0, stream>>>(
        LN, Wup, nullptr, nullptr, UP, BSROWS, 2 * INNER, DMODEL,
        DMODEL, DMODEL, 2 * INNER, 0, 0, 0, 0);
    long ne = (long)BSROWS * INNER;
    conv_silu<<<(ne + 255) / 256, 256, 0, stream>>>(UP, cw, cbp, XC, INNER, 2 * INNER, INNER);
    headwise4<<<(ne + 255) / 256, 256, 0, stream>>>(XC, qw, QKV, INNER, 3 * INNER, 0);
    headwise4<<<(ne + 255) / 256, 256, 0, stream>>>(XC, kw, QKV, INNER, 3 * INNER, INNER);
    headwise4<<<(ne + 255) / 256, 256, 0, stream>>>(UP, vw, QKV, 2 * INNER, 3 * INNER, 2 * INNER);
    igfg_kernel<<<BSROWS, 256, 0, stream>>>(QKV, igw, igb, fgw, fgb, IG, FG);
    lsig_scan<<<NBATCH * NHEAD, 1024, 0, stream>>>(FG, LFC);
    mlstm_attn<<<dim3(SEQ / 16, NHEAD, NBATCH), 128, 0, stream>>>(QKV, IG, LFC, H);
    rownorm<<<BSROWS * NHEAD, 256, 0, stream>>>(H, nw, H, DH_M, NHEAD, 1e-5f);
    combine_mlstm<<<(ne + 255) / 256, 256, 0, stream>>>(H, XC, UP, sk);
    gemm_bf16w<<<dim3(BSROWS / 64, DMODEL / 64, 1), 128, 0, stream>>>(
        H, Wd, nullptr, resIn, resOut, BSROWS, DMODEL, INNER,
        INNER, INNER, DMODEL, 0, 0, 0, 0);
  };

  auto sblock = [&](int p0) {
    const float *ln1 = F(p0 + 0), *cw = F(p0 + 1), *cbp = F(p0 + 2),
                *iw = F(p0 + 3), *fw = F(p0 + 4), *zw = F(p0 + 5), *ow = F(p0 + 6),
                *R = F(p0 + 7), *bb = F(p0 + 8), *gnw = F(p0 + 9),
                *ln2 = F(p0 + 10), *ffu = F(p0 + 11), *ffd = F(p0 + 12);
    rownorm<<<BSROWS, 256, 0, stream>>>(RES, ln1, LN, DMODEL, 1, 1e-5f);
    long nc = (long)BSROWS * DMODEL;
    conv_silu<<<(nc + 255) / 256, 256, 0, stream>>>(LN, cw, cbp, XC, DMODEL, DMODEL, DMODEL);
    const float* gsrc[4] = {XC, XC, LN, LN};
    const float* gw[4]   = {iw, fw, zw, ow};
    for (int g = 0; g < 4; ++g)
      gemm_bf16w<<<dim3(BSROWS / 64, DH_S / 64, NHEAD), 128, 0, stream>>>(
          gsrc[g], gw[g], nullptr, nullptr, GATES + (size_t)g * BSROWS * DMODEL,
          BSROWS, DH_S, DH_S, DMODEL, DH_S, DMODEL,
          DH_S, (long)DH_S * DH_S, 0, DH_S);
    slstm_scan<<<NHEAD, 128, 0, stream>>>(GATES, R, bb, H);
    rownorm<<<BSROWS * NHEAD, 256, 0, stream>>>(H, gnw, H, DH_S, NHEAD, 1e-5f);
    add_inplace<<<(nc + 255) / 256, 256, 0, stream>>>(RES, H, nc);
    rownorm<<<BSROWS, 256, 0, stream>>>(RES, ln2, LN, DMODEL, 1, 1e-5f);
    gemm_bf16w<<<dim3(BSROWS / 64, (2 * FFD) / 64, 1), 128, 0, stream>>>(
        LN, ffu, nullptr, nullptr, UP, BSROWS, 2 * FFD, DMODEL,
        DMODEL, DMODEL, 2 * FFD, 0, 0, 0, 0);
    long nm = (long)BSROWS * FFD;
    gelu_gate<<<(nm + 255) / 256, 256, 0, stream>>>(UP, XC);
    gemm_bf16w<<<dim3(BSROWS / 64, DMODEL / 64, 1), 128, 0, stream>>>(
        XC, ffd, nullptr, RES, RES, BSROWS, DMODEL, FFD,
        FFD, FFD, DMODEL, 0, 0, 0, 0);
  };

  mblock(1, X, RES);      // block0: residual source = input x
  sblock(15);             // block1
  mblock(28, RES, RES);   // block2

  rownorm<<<BSROWS, 256, 0, stream>>>(RES, F(42), LN, DMODEL, 1, 1e-5f);
  selu_mean<<<(NBATCH * DMODEL + 255) / 256, 256, 0, stream>>>(LN, FEAT);
  heads_kernel<<<80, 32, 0, stream>>>(FEAT, F(43), F(44), F(45), F(46), (float*)d_out);
}
